// MHA_KV_CACHE_76149770158402
// MI455X (gfx1250) — matrix operand order repaired and checked by reading, not compiled, not run on hardware
//
#include <hip/hip_runtime.h>
#include <stdint.h>

// ---------------------------------------------------------------------------
// MHA with KV cache for MI455X (gfx1250), bf16 WMMA everywhere.
// B=2, T=2048, D=1280, H=20, DH=64, S_PAST=2048, STOT=4096.
// ---------------------------------------------------------------------------

#define BDIM   2
#define TDIM   2048
#define DDIM   1280
#define HDIM   20
#define DH     64
#define S_PAST 2048
#define STOT   4096

typedef __attribute__((ext_vector_type(16))) __bf16 bf16x16;
typedef __attribute__((ext_vector_type(8)))  float  f32x8;

__device__ __forceinline__ __bf16 f2bf(float f) {
  unsigned int u = __builtin_bit_cast(unsigned int, f);
  unsigned int r = (u + 0x7FFFu + ((u >> 16) & 1u)) >> 16;   // round-to-nearest-even
  return __builtin_bit_cast(__bf16, (unsigned short)r);
}

__device__ __forceinline__ f32x8 wmma_bf16(bf16x16 a, bf16x16 b, f32x8 c) {
  // D = A(16x32 bf16) * B(32x16 bf16) + C(16x16 f32)
  return __builtin_amdgcn_wmma_f32_16x16x32_bf16(false, a, false, b, (short)0, c,
                                                 false, false);
}

// CDNA5 async global->LDS copy (16B per lane), tracked by ASYNCcnt.
// LDS operand: low 32 bits of the flat shared address == LDS byte offset
// (ISA 10.2 aperture table).
__device__ __forceinline__ void async_copy_b128(void* lds_dst, const void* gsrc) {
  unsigned lds = (unsigned)(uintptr_t)lds_dst;
  unsigned long long g = (unsigned long long)(uintptr_t)gsrc;
  asm volatile("global_load_async_to_lds_b128 %0, %1, off"
               :: "v"(lds), "v"(g) : "memory");
}
__device__ __forceinline__ void wait_async0() {
  asm volatile("s_wait_asynccnt 0x0" ::: "memory");
}

// ---------------------------------------------------------------------------
// fp32 -> bf16 bulk convert (4 elements/thread, n divisible by 1024)
// ---------------------------------------------------------------------------
__global__ __launch_bounds__(256) void k_f32_to_bf16(const float* __restrict__ in,
                                                     __bf16* __restrict__ out, int n) {
  int i = (blockIdx.x * 256 + threadIdx.x) * 4;
  if (i + 3 < n) {
    float4 v = *reinterpret_cast<const float4*>(in + i);
    out[i + 0] = f2bf(v.x);
    out[i + 1] = f2bf(v.y);
    out[i + 2] = f2bf(v.z);
    out[i + 3] = f2bf(v.w);
  }
}

// ---------------------------------------------------------------------------
// Copy past_K/past_V [B,H,2048,64] into fp32 output slabs [B,H,4096,64]
// (rows 0..2047) and into bf16 mirrors used by the attention kernel.
// blockIdx.y: 0 -> K, 1 -> V.
// ---------------------------------------------------------------------------
__global__ __launch_bounds__(256)
void k_copy_past(const float* __restrict__ pastK, const float* __restrict__ pastV,
                 float* __restrict__ Kout, __bf16* __restrict__ Kb,
                 float* __restrict__ Vout, __bf16* __restrict__ Vb) {
  int i = blockIdx.x * 256 + threadIdx.x;            // 0 .. B*H*S_PAST*DH-1
  const int per = S_PAST * DH;                       // 131072
  int bh = i / per;
  int o  = i + bh * per;                             // destination slab has STOT rows
  if (blockIdx.y == 0) {
    float v = pastK[i];
    Kout[o] = v;  Kb[o] = f2bf(v);
  } else {
    float v = pastV[i];
    Vout[o] = v;  Vb[o] = f2bf(v);
  }
}

// ---------------------------------------------------------------------------
// bf16 GEMM, C[4096,1280] = A[4096,1280] * W[1280,1280], 128x64 block tile,
// BK=32, 256 threads = 8 waves, each wave owns a 32x32 sub-tile (2x2 WMMA).
// A tile staged via GLOBAL_LOAD_ASYNC_TO_LDS_B128; W tile transposed via VGPRs.
// mode==0: fused QKV store (grid.z selects Wq/Wk/Wv and destination).
// mode==1: plain fp32 store to Cout.
// ---------------------------------------------------------------------------
__global__ __launch_bounds__(256)
void k_gemm_bf16(const __bf16* __restrict__ A,
                 const __bf16* __restrict__ W0, const __bf16* __restrict__ W1,
                 const __bf16* __restrict__ W2, int mode,
                 __bf16* __restrict__ Qb,
                 float* __restrict__ Kout, __bf16* __restrict__ Kb,
                 float* __restrict__ Vout, __bf16* __restrict__ Vb,
                 float* __restrict__ Cout) {
  __shared__ __bf16 ldsA[128][40];   // rows padded: 40*2B = 80B (16B aligned)
  __shared__ __bf16 ldsWt[64][40];   // transposed W tile: [n][k]

  const int tid  = threadIdx.x;
  const int wave = tid >> 5, lane = tid & 31;
  const int lm   = lane & 15, lhi = lane >> 4;
  const int n0   = blockIdx.x * 64;
  const int m0   = blockIdx.y * 128;
  const __bf16* W = (blockIdx.z == 0) ? W0 : (blockIdx.z == 1 ? W1 : W2);

  const int wm = (wave >> 1) * 32;   // 0,32,64,96
  const int wn = (wave & 1) * 32;    // 0,32
  const int kbA = lhi * 16;          // A-fragment K base, contiguous k = 16*lhi + s
  const int kbB = lhi * 16;          // B-fragment K base, contiguous k = 16*lhi + s (same as A)

  f32x8 acc[2][2] = {};

  for (int k0 = 0; k0 < DDIM; k0 += 32) {
    __syncthreads();
    // A tile 128x32: 512 chunks of 8 bf16, 2 async b128 copies per thread
    {
      int c0 = tid * 2;
#pragma unroll
      for (int it = 0; it < 2; ++it) {
        int ch = c0 + it;
        int r  = ch >> 2;
        int cc = (ch & 3) * 8;
        async_copy_b128(&ldsA[r][cc], &A[(m0 + r) * DDIM + k0 + cc]);
      }
    }
    // W tile 32x64, stored transposed into LDS for contiguous B-fragment reads
    {
      int kr = tid >> 3;
      int nc = (tid & 7) * 8;
      const __bf16* src = &W[(k0 + kr) * DDIM + n0 + nc];
#pragma unroll
      for (int j = 0; j < 8; ++j) ldsWt[nc + j][kr] = src[j];
    }
    wait_async0();
    __syncthreads();

    bf16x16 afrag[2], bfrag[2];
#pragma unroll
    for (int a = 0; a < 2; ++a) {
      const __bf16* row = &ldsA[wm + 16 * a + lm][0];
#pragma unroll
      for (int i = 0; i < 8; ++i) afrag[a][i]     = row[kbA + i];
#pragma unroll
      for (int i = 0; i < 8; ++i) afrag[a][8 + i] = row[kbA + 8 + i];
    }
#pragma unroll
    for (int q = 0; q < 2; ++q) {
      const __bf16* col = &ldsWt[wn + 16 * q + lm][0];
#pragma unroll
      for (int i = 0; i < 16; ++i) bfrag[q][i] = col[kbB + i];
    }
#pragma unroll
    for (int a = 0; a < 2; ++a)
#pragma unroll
      for (int q = 0; q < 2; ++q)
        acc[a][q] = wmma_bf16(afrag[a], bfrag[q], acc[a][q]);
  }

  // epilogue: C/D layout -> m = lhi*8 + i, n = lm (ISA 7.12.2).
  // Branches hoisted out of the store loops for straight-line stores.
  const int mrow = m0 + wm + lhi * 8;   // + 16*a + i
  const int ncol = n0 + wn + lm;        // + 16*q
  if (mode == 1) {
#pragma unroll
    for (int a = 0; a < 2; ++a)
#pragma unroll
      for (int q = 0; q < 2; ++q)
#pragma unroll
        for (int i = 0; i < 8; ++i)
          Cout[(mrow + 16 * a + i) * DDIM + ncol + 16 * q] = acc[a][q][i];
  } else if (blockIdx.z == 0) {
    // Q, pre-scaled by 1/sqrt(DH), layout [B,H,T,DH]
#pragma unroll
    for (int a = 0; a < 2; ++a)
#pragma unroll
      for (int q = 0; q < 2; ++q) {
        int n = ncol + 16 * q;
        int h = n >> 6, dh = n & 63;
#pragma unroll
        for (int i = 0; i < 8; ++i) {
          int m = mrow + 16 * a + i;
          int b = m >> 11, t = m & 2047;
          Qb[((b * HDIM + h) * TDIM + t) * DH + dh] = f2bf(acc[a][q][i] * 0.125f);
        }
      }
  } else {
    float*  Fo = (blockIdx.z == 1) ? Kout : Vout;
    __bf16* Bo = (blockIdx.z == 1) ? Kb : Vb;
#pragma unroll
    for (int a = 0; a < 2; ++a)
#pragma unroll
      for (int q = 0; q < 2; ++q) {
        int n = ncol + 16 * q;
        int h = n >> 6, dh = n & 63;
#pragma unroll
        for (int i = 0; i < 8; ++i) {
          int m = mrow + 16 * a + i;
          int b = m >> 11, t = m & 2047;
          int idx = ((b * HDIM + h) * STOT + (S_PAST + t)) * DH + dh;
          float v = acc[a][q][i];
          Fo[idx] = v;
          Bo[idx] = f2bf(v);
        }
      }
  }
}

// ---------------------------------------------------------------------------
// Flash attention: grid (T/128, B*H), 256 threads = 8 waves; each wave owns a
// 16-row Q strip, streams S in 32-wide tiles, online softmax.
// K tile staged via async global->LDS; V tile transposed via VGPRs.
// Output Ob is bf16 [B*T, D] with head interleave (ready for W_o GEMM).
// ---------------------------------------------------------------------------
__global__ __launch_bounds__(256)
void k_attention(const __bf16* __restrict__ Qb, const __bf16* __restrict__ Kb,
                 const __bf16* __restrict__ Vb, __bf16* __restrict__ Ob) {
  __shared__ __bf16 ldsK[32][72];        // K tile rows: 72*2B = 144B (16B aligned)
  __shared__ __bf16 ldsVt[64][40];       // V tile transposed: [dh][s]
  __shared__ __bf16 ldsP[8][16][40];     // per-wave P relayout buffer

  const int tid  = threadIdx.x;
  const int wave = tid >> 5, lane = tid & 31;
  const int lm   = lane & 15, lhi = lane >> 4;
  const int bh   = blockIdx.y;
  const int b    = bh / HDIM, h = bh % HDIM;
  const int row0 = blockIdx.x * 128 + wave * 16;

  const __bf16* Kbase = Kb + bh * STOT * DH;
  const __bf16* Vbase = Vb + bh * STOT * DH;

  // Q fragments (A 16x32 layout), 2 halves covering dh 0..63
  bf16x16 qf[2];
  {
    const __bf16* qr = Qb + ((bh * TDIM) + row0 + lm) * DH;
#pragma unroll
    for (int half = 0; half < 2; ++half) {
#pragma unroll
      for (int i = 0; i < 8; ++i) qf[half][i]     = qr[32 * half + lhi * 16 + i];
#pragma unroll
      for (int i = 0; i < 8; ++i) qf[half][8 + i] = qr[32 * half + lhi * 16 + 8 + i];
    }
  }

  f32x8 oacc[4] = {};
  float mrun[8], lrun[8];
#pragma unroll
  for (int i = 0; i < 8; ++i) { mrun[i] = -1e30f; lrun[i] = 0.f; }

  for (int s0 = 0; s0 < STOT; s0 += 32) {
    __syncthreads();
    // K tile 32x64 async to LDS; V tile transposed [dh][s] via VGPRs
    {
      int r = tid >> 3, cc = (tid & 7) * 8;
      async_copy_b128(&ldsK[r][cc], &Kbase[(s0 + r) * DH + cc]);
      const __bf16* vs = &Vbase[(s0 + r) * DH + cc];
#pragma unroll
      for (int j = 0; j < 8; ++j) ldsVt[cc + j][r] = vs[j];
    }
    wait_async0();
    __syncthreads();

    // scores S = Q @ K^T for two 16-wide s sub-tiles
    f32x8 sacc[2] = {};
#pragma unroll
    for (int sub = 0; sub < 2; ++sub) {
      const __bf16* krow = &ldsK[sub * 16 + lm][0];
#pragma unroll
      for (int half = 0; half < 2; ++half) {
        bf16x16 kf;
#pragma unroll
        for (int i = 0; i < 16; ++i) kf[i] = krow[half * 32 + lhi * 16 + i];
        sacc[sub] = wmma_bf16(qf[half], kf, sacc[sub]);
      }
    }

    // online softmax: row max / rescale / exp / row sum (16-lane groups)
    float mtile[8];
#pragma unroll
    for (int i = 0; i < 8; ++i) mtile[i] = fmaxf(sacc[0][i], sacc[1][i]);
#pragma unroll
    for (int off = 1; off < 16; off <<= 1)
#pragma unroll
      for (int i = 0; i < 8; ++i)
        mtile[i] = fmaxf(mtile[i], __shfl_xor(mtile[i], off, 32));

    float alpha[8], rsum[8];
#pragma unroll
    for (int i = 0; i < 8; ++i) {
      float mnew = fmaxf(mrun[i], mtile[i]);
      alpha[i] = __expf(mrun[i] - mnew);
      mrun[i]  = mnew;
      float p0 = __expf(sacc[0][i] - mnew);
      float p1 = __expf(sacc[1][i] - mnew);
      sacc[0][i] = p0; sacc[1][i] = p1;
      rsum[i] = p0 + p1;
    }
#pragma unroll
    for (int off = 1; off < 16; off <<= 1)
#pragma unroll
      for (int i = 0; i < 8; ++i) rsum[i] += __shfl_xor(rsum[i], off, 32);
#pragma unroll
    for (int i = 0; i < 8; ++i) lrun[i] = lrun[i] * alpha[i] + rsum[i];
#pragma unroll
    for (int c = 0; c < 4; ++c)
#pragma unroll
      for (int i = 0; i < 8; ++i) oacc[c][i] *= alpha[i];

    // P: C-layout -> LDS -> A-fragment relayout
#pragma unroll
    for (int sub = 0; sub < 2; ++sub)
#pragma unroll
      for (int i = 0; i < 8; ++i)
        ldsP[wave][lhi * 8 + i][sub * 16 + lm] = f2bf(sacc[sub][i]);
    __syncthreads();

    bf16x16 pf;
    {
      const __bf16* pr = &ldsP[wave][lm][0];
#pragma unroll
      for (int i = 0; i < 8; ++i) pf[i]     = pr[lhi * 16 + i];
#pragma unroll
      for (int i = 0; i < 8; ++i) pf[8 + i] = pr[lhi * 16 + 8 + i];
    }
    // O += P(16x32) @ Vtile(32x64), 4 dh chunks of 16
#pragma unroll
    for (int c = 0; c < 4; ++c) {
      bf16x16 vf;
      const __bf16* vcol = &ldsVt[c * 16 + lm][0];
#pragma unroll
      for (int i = 0; i < 16; ++i) vf[i] = vcol[lhi * 16 + i];
      oacc[c] = wmma_bf16(pf, vf, oacc[c]);
    }
  }

  // normalize and store to [B*T, D] with head interleave
#pragma unroll
  for (int c = 0; c < 4; ++c)
#pragma unroll
    for (int i = 0; i < 8; ++i) {
      int t  = row0 + lhi * 8 + i;
      int dh = c * 16 + lm;
      Ob[(b * TDIM + t) * DDIM + h * DH + dh] = f2bf(oacc[c][i] / lrun[i]);
    }
}

// ---------------------------------------------------------------------------
// Host-side launcher
// ---------------------------------------------------------------------------
extern "C" void kernel_launch(void* const* d_in, const int* in_sizes, int n_in,
                              void* d_out, int out_size, void* d_ws, size_t ws_size,
                              hipStream_t stream) {
  const float* x      = (const float*)d_in[0];
  const float* past_K = (const float*)d_in[1];
  const float* past_V = (const float*)d_in[2];
  const float* w_q    = (const float*)d_in[3];
  const float* w_k    = (const float*)d_in[4];
  const float* w_v    = (const float*)d_in[5];
  const float* W_o    = (const float*)d_in[6];

  const int N_X = BDIM * TDIM * DDIM;            // 5,242,880
  const int N_W = DDIM * DDIM;                   // 1,638,400
  const int N_PAST = BDIM * HDIM * S_PAST * DH;  // 5,242,880

  float* out  = (float*)d_out;
  float* Kout = out + N_X;                       // [B,H,STOT,DH]
  float* Vout = Kout + BDIM * HDIM * STOT * DH;

  // workspace layout (bytes): ~86.5 MB total
  char* ws = (char*)d_ws;
  __bf16* xb    = (__bf16*)(ws);
  __bf16* wqb   = (__bf16*)(ws + 10485760);
  __bf16* wkb   = (__bf16*)(ws + 13762560);
  __bf16* wvb   = (__bf16*)(ws + 17039360);
  __bf16* wob   = (__bf16*)(ws + 20316160);
  __bf16* Qb    = (__bf16*)(ws + 23592960);   // [B,H,T,DH], pre-scaled
  __bf16* Kb    = (__bf16*)(ws + 34078720);   // [B,H,STOT,DH]
  __bf16* Vb    = (__bf16*)(ws + 55050240);
  __bf16* attnb = (__bf16*)(ws + 76021760);   // [B*T, D]

  // 1) fp32 -> bf16 conversions
  k_f32_to_bf16<<<N_X / 1024, 256, 0, stream>>>(x, xb, N_X);
  k_f32_to_bf16<<<N_W / 1024, 256, 0, stream>>>(w_q, wqb, N_W);
  k_f32_to_bf16<<<N_W / 1024, 256, 0, stream>>>(w_k, wkb, N_W);
  k_f32_to_bf16<<<N_W / 1024, 256, 0, stream>>>(w_v, wvb, N_W);
  k_f32_to_bf16<<<N_W / 1024, 256, 0, stream>>>(W_o, wob, N_W);

  // 2) past KV -> output slabs (fp32) + bf16 mirrors
  k_copy_past<<<dim3(N_PAST / 256, 2), 256, 0, stream>>>(past_K, past_V,
                                                         Kout, Kb, Vout, Vb);

  // 3) fused QKV projection (grid.z: 0=Q, 1=K, 2=V)
  k_gemm_bf16<<<dim3(DDIM / 64, (BDIM * TDIM) / 128, 3), 256, 0, stream>>>(
      xb, wqb, wkb, wvb, /*mode=*/0, Qb, Kout, Kb, Vout, Vb, nullptr);

  // 4) flash attention
  k_attention<<<dim3(TDIM / 128, BDIM * HDIM), 256, 0, stream>>>(Qb, Kb, Vb, attnb);

  // 5) output projection -> fp32 out
  k_gemm_bf16<<<dim3(DDIM / 64, (BDIM * TDIM) / 128, 1), 256, 0, stream>>>(
      attnb, wob, wob, wob, /*mode=*/1, nullptr, nullptr, nullptr, nullptr, nullptr,
      out);
}
